// AdaptiveActivationBlock_84250078478629
// MI455X (gfx1250) — matrix operand order repaired and checked by reading, not compiled, not run on hardware
//
#include <hip/hip_runtime.h>
#include <hip/hip_bf16.h>

// ---------------------------------------------------------------------------
// Fused deformable-conv block for MI455X (gfx1250, wave32, WMMA):
//   async(LDS) weight staging overlapped with grouped offset conv (fp32 VALU)
//   -> affine->9 offsets -> bilinear im2col into LDS (bf16)
//   -> v_wmma_f32_16x16x32_bf16 GEMM (K=288) -> BN + residual ReLU epilogue.
// im2col never touches HBM (720 MB avoided); x (80 MB) is L2-resident (192 MB).
// ---------------------------------------------------------------------------

typedef __bf16 v16bf __attribute__((ext_vector_type(16)));
typedef float  v8f   __attribute__((ext_vector_type(8)));
typedef unsigned int uint32x4 __attribute__((ext_vector_type(4)));

namespace {
constexpr int Bn  = 4;
constexpr int C   = 544;
constexpr int G   = 17;
constexpr int Cg  = 32;     // C / G
constexpr int H   = 96;
constexpr int W   = 96;
constexpr int HW  = H * W;  // 9216
constexpr int K   = Cg * 9; // 288
constexpr int LDA = 296;    // K + 8 pad (keeps 16B alignment, eases bank conflicts)
constexpr int MT  = 64;     // pixels per block (M tile)
constexpr float BN_EPS = 1e-5f;
}

__device__ __forceinline__ unsigned short f2bf(float f) {
    unsigned int u = __float_as_uint(f);
    u += 0x7FFFu + ((u >> 16) & 1u);          // round-to-nearest-even
    return (unsigned short)(u >> 16);
}

// Async global->LDS copy, 16B per lane (GLOBAL_LOAD_ASYNC_TO_LDS_B128, GV mode).
// ldsOff = low 32 bits of the generic shared-aperture address (LDS byte offset).
__device__ __forceinline__ void async_load16(unsigned ldsOff, const void* gptr) {
#if defined(__HIP_DEVICE_COMPILE__)
    unsigned long long ga = (unsigned long long)(uintptr_t)gptr;
    asm volatile("global_load_async_to_lds_b128 %0, %1, off"
                 :: "v"(ldsOff), "v"(ga) : "memory");
#endif
}
__device__ __forceinline__ void wait_asynccnt0() {
#if defined(__HIP_DEVICE_COMPILE__)
    asm volatile("s_wait_asynccnt 0x0" ::: "memory");
#endif
}

union FragU { uint32x4 u4[2]; v16bf v; };

__global__ __launch_bounds__(256)
void dcn_fused_wmma_kernel(const float* __restrict__ x,
                           const float* __restrict__ w_off,
                           const float* __restrict__ b_off,
                           const float* __restrict__ w_def,
                           const float* __restrict__ gamma,
                           const float* __restrict__ beta,
                           const float* __restrict__ rmean,
                           const float* __restrict__ rvar,
                           float* __restrict__ out) {
    __shared__ unsigned short As[MT * LDA];   // bf16 im2col tile / fp32 weight stage
    __shared__ unsigned short Bs[Cg * LDA];   // bf16 weights  [outCh][K]
    __shared__ float affS[MT * 6];            // per-pixel 2x3 affine

    const int tid     = threadIdx.x;
    const int b       = blockIdx.z;
    const int g       = blockIdx.y;
    const int pixBase = blockIdx.x * MT;

    const float* __restrict__ xg = x + ((size_t)b * C + (size_t)g * Cg) * HW;
    const float* __restrict__ wd = w_def + (size_t)(g * Cg) * K;

    // ---- Phase 0a: fire async DMA of group weight slice into LDS stage -----
    // 32*288 fp32 = 36864 B staged in the (currently unused) As region.
    {
        float* stage = (float*)As;
        for (int chunk = tid; chunk < (Cg * K) / 4; chunk += 256) {
            const unsigned ldsOff = (unsigned)(uintptr_t)(stage + chunk * 4);
            async_load16(ldsOff, wd + chunk * 4);
        }
    }

    // ---- Phase 1 (overlaps async copy): offset conv -> affS[p][0..5] -------
    for (int task = tid; task < MT * 6; task += 256) {
        const int p  = task / 6;
        const int r6 = task - p * 6;
        const int pix = pixBase + p;
        const int hh = pix / W, ww = pix - hh * W;
        const float* __restrict__ wp = w_off + (size_t)(g * 6 + r6) * K;
        float acc = b_off[g * 6 + r6];
        for (int i = 0; i < Cg; ++i) {
            const float* __restrict__ xch = xg + (size_t)i * HW;
            #pragma unroll
            for (int kh = 0; kh < 3; ++kh) {
                const int yy = hh + kh - 1;
                if (yy < 0 || yy >= H) continue;
                const float* __restrict__ row = xch + yy * W;
                #pragma unroll
                for (int kw = 0; kw < 3; ++kw) {
                    const int xx = ww + kw - 1;
                    if (xx < 0 || xx >= W) continue;
                    acc = fmaf(wp[i * 9 + kh * 3 + kw], row[xx], acc);
                }
            }
        }
        affS[task] = acc;
    }

    // ---- Phase 0b: drain async copy, convert staged fp32 weights -> bf16 ---
    wait_asynccnt0();
    __syncthreads();
    {
        const float* stage = (const float*)As;
        for (int idx = tid; idx < Cg * K; idx += 256) {
            const int o = idx / K, kk = idx - o * K;
            Bs[o * LDA + kk] = f2bf(stage[idx]);
        }
    }
    __syncthreads();

    // ---- Phase 2: bilinear-sampled im2col -> As (bf16), K index = i*9+tap --
    for (int task = tid; task < MT * 9; task += 256) {
        const int p = task / 9;
        const int k = task - p * 9;
        const float kyv = (float)(k / 3 - 1);
        const float kxv = (float)(k % 3 - 1);
        const float* a = &affS[p * 6];
        const float offy = fmaf(a[0], kyv, fmaf(a[1], kxv, a[2]));
        const float offx = fmaf(a[3], kyv, fmaf(a[4], kxv, a[5]));
        const int pix = pixBase + p;
        const int hh = pix / W, ww = pix - hh * W;
        const float py = (float)hh + kyv + offy;
        const float px = (float)ww + kxv + offx;
        const float y0f = floorf(py), x0f = floorf(px);
        const float fy = py - y0f, fx = px - x0f;
        const float w00 = (1.f - fy) * (1.f - fx), w01 = (1.f - fy) * fx;
        const float w10 = fy * (1.f - fx),         w11 = fy * fx;
        const int y0 = (int)y0f, x0 = (int)x0f, y1 = y0 + 1, x1 = x0 + 1;
        const bool vy0 = (y0 >= 0) & (y0 < H), vy1 = (y1 >= 0) & (y1 < H);
        const bool vx0 = (x0 >= 0) & (x0 < W), vx1 = (x1 >= 0) & (x1 < W);
        const int y0c = min(max(y0, 0), H - 1), y1c = min(max(y1, 0), H - 1);
        const int x0c = min(max(x0, 0), W - 1), x1c = min(max(x1, 0), W - 1);
        const int i00 = y0c * W + x0c, i01 = y0c * W + x1c;
        const int i10 = y1c * W + x0c, i11 = y1c * W + x1c;
        const float m00 = (vy0 & vx0) ? w00 : 0.f, m01 = (vy0 & vx1) ? w01 : 0.f;
        const float m10 = (vy1 & vx0) ? w10 : 0.f, m11 = (vy1 & vx1) ? w11 : 0.f;
        unsigned short* __restrict__ arow = &As[p * LDA + k];
        for (int i = 0; i < Cg; ++i) {
            const float* __restrict__ xch = xg + (size_t)i * HW;
            const float v = m00 * xch[i00] + m01 * xch[i01]
                          + m10 * xch[i10] + m11 * xch[i11];
            arow[i * 9] = f2bf(v);
        }
    }
    __syncthreads();

    // ---- Phase 3: WMMA GEMM  D[64x32] = A[64x288] * B[288x32] --------------
    const int lane  = tid & 31;
    const int wid   = tid >> 5;
    const int mBase = (wid & 3) << 4;   // 4 M tiles
    const int nBase = (wid >> 2) << 4;  // 2 N tiles
    const int half  = lane >> 4;
    const int l15   = lane & 15;

    const unsigned short* __restrict__ Arow = &As[(mBase + l15) * LDA];
    const unsigned short* __restrict__ Brow = &Bs[(nBase + l15) * LDA];

    // BN params + residual prefetch for this lane's output column
    const int oc = g * Cg + nBase + l15;
    const float inv   = rsqrtf(rvar[oc] + BN_EPS);
    const float scale = gamma[oc] * inv;
    const float shift = beta[oc] - rmean[oc] * scale;
    const float* __restrict__ resCol = x   + ((size_t)b * C + oc) * HW + pixBase + mBase;
    float*       __restrict__ outCol = out + ((size_t)b * C + oc) * HW + pixBase + mBase;
    __builtin_prefetch(resCol, 0, 1);   // global_prefetch_b8 (residual, L2-resident)

    v8f acc = {0.f, 0.f, 0.f, 0.f, 0.f, 0.f, 0.f, 0.f};
    #pragma unroll
    for (int kk = 0; kk < K; kk += 32) {
        FragU af, bf;
        // A 16x32 bf16 (ISA 7.12.2): lanes0-15 K{kk..kk+7},{kk+16..kk+23};
        //                            lanes16-31 shifted by +8.
        af.u4[0] = *(const uint32x4*)(Arow + kk + half * 8);
        af.u4[1] = *(const uint32x4*)(Arow + kk + 16 + half * 8);
        // B 32x16 bf16: lane = column N; lanes0-15 K{kk..kk+7},{kk+16..kk+23}, lanes16-31 +8.
        bf.u4[0] = *(const uint32x4*)(Brow + kk + half * 8);
        bf.u4[1] = *(const uint32x4*)(Brow + kk + 16 + half * 8);
#if defined(__HIP_DEVICE_COMPILE__)
        acc = __builtin_amdgcn_wmma_f32_16x16x32_bf16(
                  /*neg_a=*/false, af.v, /*neg_b=*/false, bf.v,
                  /*c_mod=*/(short)0, acc, /*reuse_a=*/false, /*reuse_b=*/false);
#endif
    }

    // ---- Epilogue: BN + residual ReLU, f32 C/D layout (VGPR r -> M=r/r+8) --
    // Each lane's 8 pixels are consecutive -> two float4 (b128) loads/stores.
    const int mOff = half << 3;
    const float4 r0 = *(const float4*)(resCol + mOff);
    const float4 r1 = *(const float4*)(resCol + mOff + 4);
    float4 o0, o1;
    o0.x = fmaxf(fmaf(acc[0], scale, shift) + r0.x, 0.f);
    o0.y = fmaxf(fmaf(acc[1], scale, shift) + r0.y, 0.f);
    o0.z = fmaxf(fmaf(acc[2], scale, shift) + r0.z, 0.f);
    o0.w = fmaxf(fmaf(acc[3], scale, shift) + r0.w, 0.f);
    o1.x = fmaxf(fmaf(acc[4], scale, shift) + r1.x, 0.f);
    o1.y = fmaxf(fmaf(acc[5], scale, shift) + r1.y, 0.f);
    o1.z = fmaxf(fmaf(acc[6], scale, shift) + r1.z, 0.f);
    o1.w = fmaxf(fmaf(acc[7], scale, shift) + r1.w, 0.f);
    *(float4*)(outCol + mOff)     = o0;
    *(float4*)(outCol + mOff + 4) = o1;
}

extern "C" void kernel_launch(void* const* d_in, const int* in_sizes, int n_in,
                              void* d_out, int out_size, void* d_ws, size_t ws_size,
                              hipStream_t stream) {
    (void)in_sizes; (void)n_in; (void)out_size; (void)d_ws; (void)ws_size;
    const float* x      = (const float*)d_in[0];
    const float* w_off  = (const float*)d_in[1];
    const float* b_off  = (const float*)d_in[2];
    const float* w_def  = (const float*)d_in[3];
    const float* gamma  = (const float*)d_in[4];
    const float* beta   = (const float*)d_in[5];
    const float* rmean  = (const float*)d_in[6];
    const float* rvar   = (const float*)d_in[7];
    float* out = (float*)d_out;

    dim3 grid(HW / MT, G, Bn);   // 144 x 17 x 4
    dim3 block(256);             // 8 wave32s -> 4x2 WMMA tiles of 64x32 output
    dcn_fused_wmma_kernel<<<grid, block, 0, stream>>>(
        x, w_off, b_off, w_def, gamma, beta, rmean, rvar, out);
}
